// WaveletDualAttention_55104430407786
// MI455X (gfx1250) — matrix operand order repaired and checked by reading, not compiled, not run on hardware
//
#include <hip/hip_runtime.h>
#include <hip/hip_bf16.h>

// ---------------------------------------------------------------------------
// WaveletDualAttention for MI455X (gfx1250, wave32, WMMA).
// Memory-bound problem (~54 GFLOP vs ~1.2 GB HBM traffic -> ~52us floor at
// 23.3 TB/s). GEMMs run on v_wmma_f32_16x16x32_bf16 with a hi/lo bf16 split
// (A*B ~= Ah*Bh + Ah*Bl + Al*Bh) for ~f32 accuracy at zero memory cost.
// Data operands are pre-packed into the WMMA B-fragment layout so each lane
// loads its 16 bf16 elements as one contiguous 32B chunk.
// ---------------------------------------------------------------------------

typedef __bf16 bf16_t;
typedef __attribute__((ext_vector_type(16))) __bf16 v16bf;
typedef __attribute__((ext_vector_type(8)))  float  v8f;

// Problem dims
#define BATCH 8
#define CIN   256
#define Hfull 160
#define Wfull 160
#define Hh    80
#define Wh    80
#define PIX   6400          // Hh*Wh
#define K4C   1024          // 4*CIN
#define NPT   400           // 16-pixel panels per batch image (6400/16)
#define NPT2  200           // 32-pixel tiles

// Workspace layout (bytes, all offsets 256B aligned). Total ~303 MiB.
#define OFF_WAV_HI   ((size_t)0)
#define OFF_WAV_LO   ((size_t)104857600)
#define OFF_FUSED    ((size_t)209715200)
#define OFF_SPIN     ((size_t)262144000)
#define OFF_SIG      ((size_t)262553600)
#define OFF_AVGCH    ((size_t)262758400)
#define OFF_MAXCH    ((size_t)262766592)
#define OFF_CHMAP    ((size_t)262774784)
#define OFF_CHO_HI   ((size_t)262782976)
#define OFF_CHO_LO   ((size_t)288997376)
#define OFF_WFUS_HI  ((size_t)315211776)
#define OFF_WFUS_LO  ((size_t)315736064)
#define OFF_BIAS1    ((size_t)316260352)
#define OFF_WPROJ_HI ((size_t)316261376)
#define OFF_WPROJ_LO ((size_t)316785664)
#define OFF_BIAS2    ((size_t)317309952)

__device__ __forceinline__ unsigned short f2bf(float f) {
  bf16_t t = (bf16_t)f;
  return __builtin_bit_cast(unsigned short, t);
}
__device__ __forceinline__ float bf2f(unsigned short u) {
  bf16_t t = __builtin_bit_cast(bf16_t, u);
  return (float)t;
}

// Load a v16bf fragment from one contiguous 32B chunk.
__device__ __forceinline__ v16bf frag1(const bf16_t* p) {
  union { v16bf f; uint4 q[2]; } u;
  u.q[0] = *(const uint4*)p;
  u.q[1] = *(const uint4*)(p + 8);
  return u.f;
}
// Load a v16bf fragment from two 16B chunks (A operand: K-chunks 16 apart).
__device__ __forceinline__ v16bf frag2(const bf16_t* p0, const bf16_t* p1) {
  union { v16bf f; uint4 q[2]; } u;
  u.q[0] = *(const uint4*)p0;
  u.q[1] = *(const uint4*)p1;
  return u.f;
}

// acc += (Ah+Al)*(Bh+Bl), dropping the O(2^-16) Al*Bl term.
__device__ __forceinline__ void mma3(v8f& acc, v16bf ah, v16bf al, v16bf bh, v16bf bl) {
  acc = __builtin_amdgcn_wmma_f32_16x16x32_bf16(false, ah, false, bh, (short)0, acc, false, false);
  acc = __builtin_amdgcn_wmma_f32_16x16x32_bf16(false, ah, false, bl, (short)0, acc, false, false);
  acc = __builtin_amdgcn_wmma_f32_16x16x32_bf16(false, al, false, bh, (short)0, acc, false, false);
}

// ---------------------------------------------------------------------------
// Weight fold: BN folded into conv weights, bf16 hi/lo split.
// fusion_w: (256,1024) row-major [o][c]
// ---------------------------------------------------------------------------
__global__ __launch_bounds__(256) void fold_fusion_kernel(
    const float* __restrict__ w, const float* __restrict__ cb,
    const float* __restrict__ g, const float* __restrict__ bb,
    const float* __restrict__ m, const float* __restrict__ v,
    bf16_t* __restrict__ whi, bf16_t* __restrict__ wlo, float* __restrict__ bias) {
  const int idx = blockIdx.x * 256 + threadIdx.x;      // 262144
  const int o = idx >> 10, c = idx & 1023;
  const float inv = g[o] * rsqrtf(v[o] + 1e-5f);
  const float val = w[idx] * inv;
  bf16_t h = (bf16_t)val;
  whi[idx] = h;
  wlo[idx] = (bf16_t)(val - (float)h);
  if (c == 0) bias[o] = cb[o] * inv + bb[o] - m[o] * inv;
}

// proj_w: (1024,256) row-major [o][c]
__global__ __launch_bounds__(256) void fold_proj_kernel(
    const float* __restrict__ w, const float* __restrict__ cb,
    const float* __restrict__ g, const float* __restrict__ bb,
    const float* __restrict__ m, const float* __restrict__ v,
    bf16_t* __restrict__ whi, bf16_t* __restrict__ wlo, float* __restrict__ bias) {
  const int idx = blockIdx.x * 256 + threadIdx.x;      // 262144
  const int o = idx >> 8, c = idx & 255;
  const float inv = g[o] * rsqrtf(v[o] + 1e-5f);
  const float val = w[idx] * inv;
  bf16_t h = (bf16_t)val;
  whi[idx] = h;
  wlo[idx] = (bf16_t)(val - (float)h);
  if (c == 0) bias[o] = cb[o] * inv + bb[o] - m[o] * inv;
}

// ---------------------------------------------------------------------------
// Haar DWT + bf16 hi/lo split, packed in WMMA B-fragment layout:
//   logical (b, c, p): pt=p/16, pl=p%16, kt=c/32, kk=c%32
//   lane = pl + 16*((kk/8)%2), e = 8*(kk/16) + kk%8
//   idx  = (((b*400+pt)*32 + kt)*32 + lane)*16 + e
// Each block: one (b, 16-pixel panel, 32 x-channels) -> all 4 subbands.
// x read exactly once (210 MB); coalesced 32B loads/16B stores.
// ---------------------------------------------------------------------------
__global__ __launch_bounds__(256) void dwt_pack_kernel(
    const float* __restrict__ x, bf16_t* __restrict__ wav_hi, bf16_t* __restrict__ wav_lo) {
  const int pt = blockIdx.x, ktg = blockIdx.y, b = blockIdx.z;
  const int tid = threadIdx.x;
  __shared__ float s_x[2048];                 // [cc_local*2+row][32 cols]
  const int i = pt / 5, jt = pt - i * 5;      // panel -> (row i, col-group jt)
  {
    const int seg = tid >> 2;                 // 0..63  (cc_local, quad-row)
    const int off = (tid & 3) * 8;
    const int ccl = seg >> 1, row = seg & 1;
    const float* src = x + ((long)(b * 256 + ktg * 32 + ccl) * 160 + (2 * i + row)) * 160
                         + jt * 32 + off;
    float4 v0 = *(const float4*)src;
    float4 v1 = *(const float4*)(src + 4);
    float* d = &s_x[seg * 32 + off];
    *(float4*)d = v0;
    *(float4*)(d + 4) = v1;
  }
  __syncthreads();
  const int s = tid >> 6;                     // subband 0..3
  const int lane = (tid >> 1) & 31;
  const int half = tid & 1;
  const int lhh = lane >> 4, lm = lane & 15;
  const float fb = (s & 2) ? -0.5f : 0.5f;
  const float fc = (s & 1) ? -0.5f : 0.5f;
  const float fd = (((s & 1) ^ (s >> 1)) & 1) ? -0.5f : 0.5f;
  union { unsigned short us[8]; uint4 q; } oh, ol;
#pragma unroll
  for (int e = 0; e < 8; ++e) {
    const int kk = half * 16 + e + lhh * 8;   // B-fragment K index
    const float* qp = &s_x[(kk * 2) * 32 + 2 * lm];
    const float a = qp[0], bv = qp[1], cv = qp[32], dv = qp[33];
    const float val = 0.5f * a + fb * bv + fc * cv + fd * dv;
    const unsigned short h = f2bf(val);
    oh.us[e] = h;
    ol.us[e] = f2bf(val - bf2f(h));
  }
  const long fidx = ((long)((b * 400 + pt) * 32) + (s * 8 + ktg)) * 512
                  + (long)lane * 16 + half * 8;
  *(uint4*)(wav_hi + fidx) = oh.q;
  *(uint4*)(wav_lo + fidx) = ol.q;
}

// ---------------------------------------------------------------------------
// Fusion GEMM: fused[b,o,p] = relu(sum_c wfold[o,c]*wav[b,c,p] + bias1[o])
// M=256 (all channels in one WG so channel mean/max fall out), N=32, K=1024.
// 8 waves, each 2 mtiles x 2 ntiles, 12 v_wmma per K-step.
// ---------------------------------------------------------------------------
__global__ __launch_bounds__(256) void fusion_gemm_kernel(
    const bf16_t* __restrict__ wav_hi, const bf16_t* __restrict__ wav_lo,
    const bf16_t* __restrict__ wf_hi, const bf16_t* __restrict__ wf_lo,
    const float* __restrict__ bias1,
    float* __restrict__ fused, float* __restrict__ sp_in) {
  const int b = blockIdx.y, pt2 = blockIdx.x;
  const int tid = threadIdx.x;
  const int wave = tid >> 5, lane = tid & 31;
  const int lh = lane >> 4, lm = lane & 15;

  __shared__ float    s_bias[256];
  __shared__ float    s_sum[32];
  __shared__ unsigned s_max[32];
  s_bias[tid] = bias1[tid];
  if (tid < 32) { s_sum[tid] = 0.0f; s_max[tid] = 0u; }
  __syncthreads();

  v8f acc00 = {}, acc01 = {}, acc10 = {}, acc11 = {};
  const int mt0 = wave * 2;
  const long arow0 = (long)(mt0 * 16 + lm) * 1024;
  const long arow1 = (long)(mt0 * 16 + 16 + lm) * 1024;
  const long bbase0 = (long)(b * 400 + pt2 * 2 + 0) * 32 * 512 + (long)lane * 16;
  const long bbase1 = (long)(b * 400 + pt2 * 2 + 1) * 32 * 512 + (long)lane * 16;

  for (int kt = 0; kt < 32; ++kt) {
    const int cb = kt * 32 + lh * 8;          // A-operand K chunk base
    v16bf a0h = frag2(wf_hi + arow0 + cb, wf_hi + arow0 + cb + 16);
    v16bf a0l = frag2(wf_lo + arow0 + cb, wf_lo + arow0 + cb + 16);
    v16bf a1h = frag2(wf_hi + arow1 + cb, wf_hi + arow1 + cb + 16);
    v16bf a1l = frag2(wf_lo + arow1 + cb, wf_lo + arow1 + cb + 16);
    const long bo = (long)kt * 512;
    v16bf b0h = frag1(wav_hi + bbase0 + bo);
    v16bf b0l = frag1(wav_lo + bbase0 + bo);
    v16bf b1h = frag1(wav_hi + bbase1 + bo);
    v16bf b1l = frag1(wav_lo + bbase1 + bo);
    mma3(acc00, a0h, a0l, b0h, b0l);
    mma3(acc01, a0h, a0l, b1h, b1l);
    mma3(acc10, a1h, a1l, b0h, b0l);
    mma3(acc11, a1h, a1l, b1h, b1l);
  }

  const long fbse = (long)b * 256 * 6400 + (long)pt2 * 32;
  v8f* accs[4] = { &acc00, &acc01, &acc10, &acc11 };
#pragma unroll
  for (int mi = 0; mi < 2; ++mi)
#pragma unroll
    for (int nt = 0; nt < 2; ++nt) {
      v8f& A = *accs[mi * 2 + nt];
      const int pl = nt * 16 + lm;
      float ps = 0.0f, pm = 0.0f;
#pragma unroll
      for (int v = 0; v < 8; ++v) {
        const int row = (mt0 + mi) * 16 + v + lh * 8;   // D-layout: M = v + 8*(lane/16)
        float r = fmaxf(A[v] + s_bias[row], 0.0f);
        fused[fbse + (long)row * 6400 + pl] = r;
        ps += r;
        pm = fmaxf(pm, r);
      }
      atomicAdd(&s_sum[pl], ps);                         // ds_add_f32
      atomicMax(&s_max[pl], __float_as_uint(pm));        // non-neg floats: uint order ok
    }
  __syncthreads();
  if (tid < 32) {
    const long o = (long)b * 2 * 6400 + (long)pt2 * 32 + tid;
    sp_in[o] = s_sum[tid] * (1.0f / 256.0f);
    sp_in[o + 6400] = __uint_as_float(s_max[tid]);
  }
}

// ---------------------------------------------------------------------------
// 7x7 spatial conv (2->1 ch) + sigmoid. Tiny.
// ---------------------------------------------------------------------------
__global__ __launch_bounds__(256) void spatial_conv_kernel(
    const float* __restrict__ sp_in, const float* __restrict__ sa_w,
    float* __restrict__ sig) {
  __shared__ float w[98];
  const int tid = threadIdx.x;
  if (tid < 98) w[tid] = sa_w[tid];
  __syncthreads();
  const int idx = blockIdx.x * 256 + tid;     // 51200
  const int b = idx / 6400, p = idx % 6400;
  const int i = p / 80, j = p - i * 80;
  float acc = 0.0f;
  for (int ic = 0; ic < 2; ++ic) {
    const float* base = sp_in + ((long)b * 2 + ic) * 6400;
    for (int ky = 0; ky < 7; ++ky) {
      const int y = i + ky - 3;
      if (y < 0 || y >= 80) continue;
      for (int kx = 0; kx < 7; ++kx) {
        const int xx = j + kx - 3;
        if (xx < 0 || xx >= 80) continue;
        acc += base[y * 80 + xx] * w[ic * 49 + ky * 7 + kx];
      }
    }
  }
  sig[idx] = 1.0f / (1.0f + __expf(-acc));
}

// ---------------------------------------------------------------------------
// Per-(b,c) global mean/max of spatial_out = sig * fused. One block per (b,c).
// ---------------------------------------------------------------------------
__global__ __launch_bounds__(256) void pool_kernel(
    const float* __restrict__ fused, const float* __restrict__ sig,
    float* __restrict__ avgch, float* __restrict__ maxch) {
  const int bc = blockIdx.x;                  // 2048
  const int b = bc >> 8;
  const float* f = fused + (long)bc * 6400;
  const float* s = sig + (long)b * 6400;
  const int tid = threadIdx.x;
  float sum = 0.0f, mx = 0.0f;                // spatial_out >= 0
  for (int k = tid; k < 6400; k += 256) {
    const float v = f[k] * s[k];
    sum += v;
    mx = fmaxf(mx, v);
  }
  __shared__ float ssum[256];
  __shared__ float smax[256];
  ssum[tid] = sum; smax[tid] = mx;
  __syncthreads();
  for (int st = 128; st > 0; st >>= 1) {
    if (tid < st) {
      ssum[tid] += ssum[tid + st];
      smax[tid] = fmaxf(smax[tid], smax[tid + st]);
    }
    __syncthreads();
  }
  if (tid == 0) {
    avgch[bc] = ssum[0] * (1.0f / 6400.0f);
    maxch[bc] = smax[0];
  }
}

// ---------------------------------------------------------------------------
// Channel-attention MLP: chmap = sigmoid(w2@relu(w1@avg) + w2@relu(w1@max))
// ---------------------------------------------------------------------------
__global__ __launch_bounds__(256) void mlp_kernel(
    const float* __restrict__ avgch, const float* __restrict__ maxch,
    const float* __restrict__ w1, const float* __restrict__ w2,
    float* __restrict__ chmap) {
  const int b = blockIdx.x, tid = threadIdx.x;
  __shared__ float h[16];
  if (tid < 16) {
    float sa = 0.0f, sm = 0.0f;
    const float* wr = w1 + tid * 256;
    const float* pa = avgch + b * 256;
    const float* pm = maxch + b * 256;
    for (int c = 0; c < 256; ++c) { sa += wr[c] * pa[c]; sm += wr[c] * pm[c]; }
    h[tid] = fmaxf(sa, 0.0f) + fmaxf(sm, 0.0f);
  }
  __syncthreads();
  float v = 0.0f;
  const float* w2r = w2 + tid * 16;
#pragma unroll
  for (int j = 0; j < 16; ++j) v += w2r[j] * h[j];
  chmap[b * 256 + tid] = 1.0f / (1.0f + __expf(-v));
}

// ---------------------------------------------------------------------------
// channel_out = chmap[b,c]*sig[b,p]*fused[b,c,p] -> bf16 hi/lo B-fragments.
// idx = (((b*400+pt)*8 + kt)*32 + lane)*16 + e   (K dim here is 256 -> kt 0..7)
// ---------------------------------------------------------------------------
__global__ __launch_bounds__(256) void chan_pack_kernel(
    const float* __restrict__ fused, const float* __restrict__ sig,
    const float* __restrict__ chmap,
    bf16_t* __restrict__ cho_hi, bf16_t* __restrict__ cho_lo) {
  const int pt = blockIdx.x, b = blockIdx.z;
  const int tid = threadIdx.x;
  const int kt = blockIdx.y * 4 + (tid >> 6);
  const int lane = (tid >> 1) & 31;
  const int half = tid & 1;
  const int lhh = lane >> 4, lm = lane & 15;
  const int p = pt * 16 + lm;
  const float sg = sig[(long)b * 6400 + p];
  union { unsigned short us[8]; uint4 q; } oh, ol;
#pragma unroll
  for (int e = 0; e < 8; ++e) {
    const int kk = half * 16 + e + lhh * 8;
    const int c = kt * 32 + kk;
    const float val = fused[((long)(b * 256 + c)) * 6400 + p] * sg * chmap[b * 256 + c];
    const unsigned short h = f2bf(val);
    oh.us[e] = h;
    ol.us[e] = f2bf(val - bf2f(h));
  }
  const long fidx = ((long)((b * 400 + pt) * 8) + kt) * 512 + (long)lane * 16 + half * 8;
  *(uint4*)(cho_hi + fidx) = oh.q;
  *(uint4*)(cho_lo + fidx) = ol.q;
}

// ---------------------------------------------------------------------------
// Proj GEMM + IDWT + residual. WG owns 4 subbands x 64 cc (the rows IDWT
// needs): M=256 gathered rows of proj_w, K=256, N=32 pixels. Epilogue stages
// relu(projected) tiles in LDS, recombines Haar quads, adds identity.
// ---------------------------------------------------------------------------
__global__ __launch_bounds__(256) void proj_gemm_idwt_kernel(
    const bf16_t* __restrict__ cho_hi, const bf16_t* __restrict__ cho_lo,
    const bf16_t* __restrict__ wp_hi, const bf16_t* __restrict__ wp_lo,
    const float* __restrict__ bias2,
    const float* __restrict__ x, float* __restrict__ out) {
  const int b = blockIdx.z;
  const int cc0 = blockIdx.y * 64;
  const int pt2 = blockIdx.x;
  const int tid = threadIdx.x;
  const int wave = tid >> 5, lane = tid & 31;
  const int lh = lane >> 4, lm = lane & 15;

  __shared__ float s_bias[256];
  __shared__ float s_tile[16 * 16 * 32];      // 32 KB: [mt=s*4+q][row 16][pix 32]
  {
    const int mt = tid >> 4, r = tid & 15;    // mt -> global row (s*256 + cc0 + q*16 + r)
    s_bias[tid] = bias2[(mt >> 2) * 256 + cc0 + (mt & 3) * 16 + r];
  }
  __syncthreads();

  v8f acc00 = {}, acc01 = {}, acc10 = {}, acc11 = {};
  const int mtA = wave * 2, mtB = wave * 2 + 1;
  const long arow0 = (long)((mtA >> 2) * 256 + cc0 + (mtA & 3) * 16 + lm) * 256;
  const long arow1 = (long)((mtB >> 2) * 256 + cc0 + (mtB & 3) * 16 + lm) * 256;
  const long bbase0 = (long)(b * 400 + pt2 * 2 + 0) * 8 * 512 + (long)lane * 16;
  const long bbase1 = (long)(b * 400 + pt2 * 2 + 1) * 8 * 512 + (long)lane * 16;

  for (int kt = 0; kt < 8; ++kt) {
    const int cb = kt * 32 + lh * 8;
    v16bf a0h = frag2(wp_hi + arow0 + cb, wp_hi + arow0 + cb + 16);
    v16bf a0l = frag2(wp_lo + arow0 + cb, wp_lo + arow0 + cb + 16);
    v16bf a1h = frag2(wp_hi + arow1 + cb, wp_hi + arow1 + cb + 16);
    v16bf a1l = frag2(wp_lo + arow1 + cb, wp_lo + arow1 + cb + 16);
    const long bo = (long)kt * 512;
    v16bf b0h = frag1(cho_hi + bbase0 + bo);
    v16bf b0l = frag1(cho_lo + bbase0 + bo);
    v16bf b1h = frag1(cho_hi + bbase1 + bo);
    v16bf b1l = frag1(cho_lo + bbase1 + bo);
    mma3(acc00, a0h, a0l, b0h, b0l);
    mma3(acc01, a0h, a0l, b1h, b1l);
    mma3(acc10, a1h, a1l, b0h, b0l);
    mma3(acc11, a1h, a1l, b1h, b1l);
  }

  v8f* accs[4] = { &acc00, &acc01, &acc10, &acc11 };
#pragma unroll
  for (int mi = 0; mi < 2; ++mi)
#pragma unroll
    for (int nt = 0; nt < 2; ++nt) {
      v8f& A = *accs[mi * 2 + nt];
      const int mt = wave * 2 + mi;
#pragma unroll
      for (int v = 0; v < 8; ++v) {
        const int r = v + lh * 8;
        s_tile[(mt * 16 + r) * 32 + nt * 16 + lm] =
            fmaxf(A[v] + s_bias[mt * 16 + r], 0.0f);
      }
    }
  __syncthreads();

#pragma unroll
  for (int it = 0; it < 8; ++it) {
    const int idx = it * 256 + tid;           // 0..2047 = 64 cc x 32 pix
    const int ccl = idx >> 5, p = idx & 31;
    const int q = ccl >> 4, r = ccl & 15;
    const float vll = s_tile[((q) * 16 + r) * 32 + p];
    const float vlh = s_tile[((4 + q) * 16 + r) * 32 + p];
    const float vhl = s_tile[((8 + q) * 16 + r) * 32 + p];
    const float vhh = s_tile[((12 + q) * 16 + r) * 32 + p];
    const float A0 = (vll + vlh + vhl + vhh) * 0.5f;
    const float B0 = (vll + vlh - vhl - vhh) * 0.5f;
    const float C0 = (vll - vlh + vhl - vhh) * 0.5f;
    const float D0 = (vll - vlh - vhl + vhh) * 0.5f;
    const int pg = pt2 * 32 + p;
    const int i = pg / 80, j = pg - i * 80;
    const long ob = (((long)(b * 256 + cc0 + ccl)) * 160 + 2 * i) * 160 + 2 * j;
    out[ob]       = A0 + x[ob];
    out[ob + 1]   = B0 + x[ob + 1];
    out[ob + 160] = C0 + x[ob + 160];
    out[ob + 161] = D0 + x[ob + 161];
  }
}

// ---------------------------------------------------------------------------
extern "C" void kernel_launch(void* const* d_in, const int* in_sizes, int n_in,
                              void* d_out, int out_size, void* d_ws, size_t ws_size,
                              hipStream_t stream) {
  (void)in_sizes; (void)n_in; (void)out_size; (void)ws_size;

  const float* x        = (const float*)d_in[0];
  const float* fusion_w = (const float*)d_in[1];
  const float* fusion_b = (const float*)d_in[2];
  const float* bn1_g    = (const float*)d_in[3];
  const float* bn1_b    = (const float*)d_in[4];
  const float* bn1_m    = (const float*)d_in[5];
  const float* bn1_v    = (const float*)d_in[6];
  const float* sa_w     = (const float*)d_in[7];
  const float* ca_w1    = (const float*)d_in[8];
  const float* ca_w2    = (const float*)d_in[9];
  const float* proj_w   = (const float*)d_in[10];
  const float* proj_b   = (const float*)d_in[11];
  const float* bn2_g    = (const float*)d_in[12];
  const float* bn2_b    = (const float*)d_in[13];
  const float* bn2_m    = (const float*)d_in[14];
  const float* bn2_v    = (const float*)d_in[15];
  float* out = (float*)d_out;

  char* ws = (char*)d_ws;
  bf16_t* wav_hi  = (bf16_t*)(ws + OFF_WAV_HI);
  bf16_t* wav_lo  = (bf16_t*)(ws + OFF_WAV_LO);
  float*  fused   = (float*)(ws + OFF_FUSED);
  float*  sp_in   = (float*)(ws + OFF_SPIN);
  float*  sig     = (float*)(ws + OFF_SIG);
  float*  avgch   = (float*)(ws + OFF_AVGCH);
  float*  maxch   = (float*)(ws + OFF_MAXCH);
  float*  chmap   = (float*)(ws + OFF_CHMAP);
  bf16_t* cho_hi  = (bf16_t*)(ws + OFF_CHO_HI);
  bf16_t* cho_lo  = (bf16_t*)(ws + OFF_CHO_LO);
  bf16_t* wfus_hi = (bf16_t*)(ws + OFF_WFUS_HI);
  bf16_t* wfus_lo = (bf16_t*)(ws + OFF_WFUS_LO);
  float*  bias1   = (float*)(ws + OFF_BIAS1);
  bf16_t* wprj_hi = (bf16_t*)(ws + OFF_WPROJ_HI);
  bf16_t* wprj_lo = (bf16_t*)(ws + OFF_WPROJ_LO);
  float*  bias2   = (float*)(ws + OFF_BIAS2);

  fold_fusion_kernel<<<1024, 256, 0, stream>>>(fusion_w, fusion_b, bn1_g, bn1_b,
                                               bn1_m, bn1_v, wfus_hi, wfus_lo, bias1);
  fold_proj_kernel<<<1024, 256, 0, stream>>>(proj_w, proj_b, bn2_g, bn2_b,
                                             bn2_m, bn2_v, wprj_hi, wprj_lo, bias2);
  dwt_pack_kernel<<<dim3(NPT, 8, BATCH), 256, 0, stream>>>(x, wav_hi, wav_lo);
  fusion_gemm_kernel<<<dim3(NPT2, BATCH), 256, 0, stream>>>(
      wav_hi, wav_lo, wfus_hi, wfus_lo, bias1, fused, sp_in);
  spatial_conv_kernel<<<200, 256, 0, stream>>>(sp_in, sa_w, sig);
  pool_kernel<<<BATCH * 256, 256, 0, stream>>>(fused, sig, avgch, maxch);
  mlp_kernel<<<BATCH, 256, 0, stream>>>(avgch, maxch, ca_w1, ca_w2, chmap);
  chan_pack_kernel<<<dim3(NPT, 2, BATCH), 256, 0, stream>>>(
      fused, sig, chmap, cho_hi, cho_lo);
  proj_gemm_idwt_kernel<<<dim3(NPT2, 4, BATCH), 256, 0, stream>>>(
      cho_hi, cho_lo, wprj_hi, wprj_lo, bias2, x, out);
}
